// CLG_70042326663848
// MI455X (gfx1250) — matrix operand order repaired and checked by reading, not compiled, not run on hardware
//
#include <hip/hip_runtime.h>

#define DIM 128

typedef __attribute__((ext_vector_type(16))) __bf16 v16bf;
typedef __attribute__((ext_vector_type(8)))  __bf16 v8bf;
typedef __attribute__((ext_vector_type(8)))  float  v8f;

__device__ __forceinline__ __bf16 f2bf(float f) { return (__bf16)f; }

// ---------------------------------------------------------------------------
// Degree count: deg[dst] += 1  (f32 atomics, exact for counts < 2^24)
// ---------------------------------------------------------------------------
__global__ __launch_bounds__(256) void k_deg(const long long* __restrict__ dst,
                                             float* __restrict__ deg, int E) {
    int e = blockIdx.x * blockDim.x + threadIdx.x;
    if (e < E) unsafeAtomicAdd(&deg[(int)dst[e]], 1.0f);
}

// dinv[i] = rsqrt(deg[i] + 1)   (+1 = self loop), in place
__global__ __launch_bounds__(256) void k_dinv(float* __restrict__ deg, int N) {
    int i = blockIdx.x * blockDim.x + threadIdx.x;
    if (i < N) deg[i] = rsqrtf(deg[i] + 1.0f);
}

// ---------------------------------------------------------------------------
// C[N x 128] = A[N x 128] @ W[128 x 128], fp32 in/out, bf16x3 WMMA compute.
// Block: 256 threads = 8 waves. Block owns 16 rows; wave w owns cols [16w,16w+16).
// Three independent accumulator chains (hi*hi, hi*lo, lo*hi) to hide the
// bf16 WMMA->WMMA RAW hazard (5 NOP slots) by interleaving.
// ---------------------------------------------------------------------------
__global__ __launch_bounds__(256) void k_gemm128(const float* __restrict__ A,
                                                 const float* __restrict__ W,
                                                 float* __restrict__ C, int nrows) {
    __shared__ __align__(32) __bf16 Ahi[16][DIM];
    __shared__ __align__(32) __bf16 Alo[16][DIM];
    __shared__ __align__(32) __bf16 WThi[DIM][DIM];   // transposed: WT[n][k] = W[k][n]
    __shared__ __align__(32) __bf16 WTlo[DIM][DIM];

    const int t      = threadIdx.x;
    const int rowblk = blockIdx.x * 16;

    // Stage W transposed as bf16 hi/lo (16384 elements, 64 per thread)
    #pragma unroll
    for (int i = 0; i < 64; ++i) {
        int idx = t + i * 256;
        int k = idx >> 7, n = idx & 127;
        float w = W[idx];
        __bf16 hi = f2bf(w);
        WThi[n][k] = hi;
        WTlo[n][k] = f2bf(w - (float)hi);
    }
    // Stage 16-row A slab as bf16 hi/lo (2048 elements, 8 per thread)
    #pragma unroll
    for (int i = 0; i < 8; ++i) {
        int idx = t + i * 256;
        int r = idx >> 7, k = idx & 127;
        int gr = rowblk + r;
        float a = (gr < nrows) ? A[(size_t)gr * DIM + k] : 0.0f;
        __bf16 hi = f2bf(a);
        Ahi[r][k] = hi;
        Alo[r][k] = f2bf(a - (float)hi);
    }
    __syncthreads();

    const int wave    = t >> 5;
    const int lane    = t & 31;
    const int l16     = lane & 15;
    const int hsel    = lane >> 4;       // which 16-lane half
    const int colbase = wave * 16;

    v8f acc0 = {};   // hi * hi
    v8f acc1 = {};   // hi * lo
    v8f acc2 = {};   // lo * hi
    #pragma unroll
    for (int kk = 0; kk < 4; ++kk) {     // K = 128 in steps of 32
        // A fragment: lane=M=l16; v0..7 = K ka..ka+15 (contiguous 16 bf16)
        int ka = kk * 32 + hsel * 16;
        v8bf a0h = *(const v8bf*)&Ahi[l16][ka];
        v8bf a1h = *(const v8bf*)&Ahi[l16][ka + 8];
        v8bf a0l = *(const v8bf*)&Alo[l16][ka];
        v8bf a1l = *(const v8bf*)&Alo[l16][ka + 8];
        v16bf ah = __builtin_shufflevector(a0h, a1h, 0,1,2,3,4,5,6,7,8,9,10,11,12,13,14,15);
        v16bf al = __builtin_shufflevector(a0l, a1l, 0,1,2,3,4,5,6,7,8,9,10,11,12,13,14,15);

        // B fragment: lane=N=l16; VGPR i holds K pair (kb+2i, kb+2i+1) — contiguous 16 bf16
        int kb = kk * 32 + hsel * 16;
        v16bf bh = *(const v16bf*)&WThi[colbase + l16][kb];
        v16bf bl = *(const v16bf*)&WTlo[colbase + l16][kb];

        acc0 = __builtin_amdgcn_wmma_f32_16x16x32_bf16(false, ah, false, bh, (short)0, acc0, false, false);
        acc1 = __builtin_amdgcn_wmma_f32_16x16x32_bf16(false, ah, false, bl, (short)0, acc1, false, false);
        acc2 = __builtin_amdgcn_wmma_f32_16x16x32_bf16(false, al, false, bh, (short)0, acc2, false, false);
    }

    // C/D layout: VGPR r -> row = r + 8*hsel, col = colbase + l16
    #pragma unroll
    for (int r = 0; r < 8; ++r) {
        int gr = rowblk + hsel * 8 + r;
        if (gr < nrows)
            C[(size_t)gr * DIM + colbase + l16] = acc0[r] + (acc1[r] + acc2[r]);
    }
}

// ---------------------------------------------------------------------------
// Self-loop init: out[i][:] = h[i][:] * dinv[i]^2   (initializes agg buffer)
// One thread per float4; n4 = N*32
// ---------------------------------------------------------------------------
__global__ __launch_bounds__(256) void k_self(const float* __restrict__ h,
                                              const float* __restrict__ dinv,
                                              float* __restrict__ out, int n4) {
    int idx = blockIdx.x * blockDim.x + threadIdx.x;
    if (idx >= n4) return;
    int node = idx >> 5;
    float s = dinv[node]; s *= s;
    float4 v = ((const float4*)h)[idx];
    float4 r; r.x = v.x * s; r.y = v.y * s; r.z = v.z * s; r.w = v.w * s;
    ((float4*)out)[idx] = r;
}

// ---------------------------------------------------------------------------
// Edge aggregation: 32 edges per wave32. Each lane loads ITS OWN edge's
// (src,dst) and precomputes norm (coalesced), prefetches its source row,
// then a 32-step loop broadcasts (s,d,norm) with __shfl; the whole wave
// moves one 512B row per step: float4 gather + 4 hardware f32 atomics/lane.
// ---------------------------------------------------------------------------
__global__ __launch_bounds__(256) void k_edge(const float* __restrict__ h,
                                              const long long* __restrict__ src,
                                              const long long* __restrict__ dst,
                                              const float* __restrict__ dinv,
                                              float* __restrict__ out, int E) {
    int wavebase = ((blockIdx.x * blockDim.x + threadIdx.x) >> 5) << 5; // first edge
    int lane     = threadIdx.x & 31;
    if (wavebase >= E) return;

    int   sv = 0, dv = 0;
    float nv = 0.0f;
    int   e  = wavebase + lane;
    if (e < E) {
        sv = (int)src[e];
        dv = (int)dst[e];
        nv = dinv[sv] * dinv[dv];
        __builtin_prefetch(h + (size_t)sv * DIM, 0, 1);   // pull source row toward L0/L2
    }

    int cnt = E - wavebase;
    if (cnt > 32) cnt = 32;

    for (int j = 0; j < cnt; ++j) {
        int   s    = __shfl(sv, j);
        int   d    = __shfl(dv, j);
        float norm = __shfl(nv, j);
        float4 v = ((const float4*)(h + (size_t)s * DIM))[lane];
        float* o = out + (size_t)d * DIM + lane * 4;
        unsafeAtomicAdd(o + 0, v.x * norm);
        unsafeAtomicAdd(o + 1, v.y * norm);
        unsafeAtomicAdd(o + 2, v.z * norm);
        unsafeAtomicAdd(o + 3, v.w * norm);
    }
}

// ---------------------------------------------------------------------------
// In-place bias + ReLU over [N x 128]; one thread per float4; n4 = N*32
// ---------------------------------------------------------------------------
__global__ __launch_bounds__(256) void k_bias_relu(float* __restrict__ x,
                                                   const float* __restrict__ b, int n4) {
    int idx = blockIdx.x * blockDim.x + threadIdx.x;
    if (idx >= n4) return;
    float4 bb = ((const float4*)b)[idx & 31];
    float4 v  = ((float4*)x)[idx];
    v.x = fmaxf(v.x + bb.x, 0.0f);
    v.y = fmaxf(v.y + bb.y, 0.0f);
    v.z = fmaxf(v.z + bb.z, 0.0f);
    v.w = fmaxf(v.w + bb.w, 0.0f);
    ((float4*)x)[idx] = v;
}

// ---------------------------------------------------------------------------
// Head: out[n][0..1] = h[n][:] @ Wl[128x2] + bl.  One wave32 per node.
// ---------------------------------------------------------------------------
__global__ __launch_bounds__(256) void k_head(const float* __restrict__ h,
                                              const float* __restrict__ Wl,
                                              const float* __restrict__ bl,
                                              float* __restrict__ out, int N) {
    int wid  = (blockIdx.x * blockDim.x + threadIdx.x) >> 5;
    int lane = threadIdx.x & 31;
    if (wid >= N) return;
    float4 v = ((const float4*)(h + (size_t)wid * DIM))[lane];
    // Wl flat: (k,j) at k*2+j. float4 q covers rows k=2q,2q+1.
    const float4* w4 = (const float4*)Wl;
    float4 wa = w4[lane * 2];       // rows 4L, 4L+1
    float4 wb = w4[lane * 2 + 1];   // rows 4L+2, 4L+3
    float p0 = v.x * wa.x + v.y * wa.z + v.z * wb.x + v.w * wb.z;
    float p1 = v.x * wa.y + v.y * wa.w + v.z * wb.y + v.w * wb.w;
    #pragma unroll
    for (int off = 16; off > 0; off >>= 1) {
        p0 += __shfl_xor(p0, off);
        p1 += __shfl_xor(p1, off);
    }
    if (lane == 0) {
        out[(size_t)wid * 2 + 0] = p0 + bl[0];
        out[(size_t)wid * 2 + 1] = p1 + bl[1];
    }
}

// ---------------------------------------------------------------------------
extern "C" void kernel_launch(void* const* d_in, const int* in_sizes, int n_in,
                              void* d_out, int out_size, void* d_ws, size_t ws_size,
                              hipStream_t stream) {
    const float*     x  = (const float*)d_in[0];
    const long long* ei = (const long long*)d_in[1];
    const float*     W1 = (const float*)d_in[2];
    const float*     b1 = (const float*)d_in[3];
    const float*     W2 = (const float*)d_in[4];
    const float*     b2 = (const float*)d_in[5];
    const float*     Wl = (const float*)d_in[6];
    const float*     bl = (const float*)d_in[7];

    const int N = in_sizes[0] / DIM;       // 100000
    const int E = in_sizes[1] / 2;         // 3200000
    const long long* src = ei;
    const long long* dst = ei + E;

    // Workspace carve: dinv (N f32) | bufh (N*128 f32) | bufa (N*128 f32)
    char*  ws        = (char*)d_ws;
    size_t degBytes  = (((size_t)N * 4) + 255) & ~(size_t)255;
    size_t featBytes = (size_t)N * DIM * 4;
    float* dinv = (float*)ws;
    float* bufh = (float*)(ws + degBytes);
    float* bufa = (float*)(ws + degBytes + featBytes);

    const int n4 = N * (DIM / 4);          // float4 count per feature matrix

    // Degrees + D^{-1/2}
    hipMemsetAsync(dinv, 0, (size_t)N * 4, stream);
    k_deg<<<(E + 255) / 256, 256, 0, stream>>>(dst, dinv, E);
    k_dinv<<<(N + 255) / 256, 256, 0, stream>>>(dinv, N);

    // Edge-aggregation grid: one wave (32 lanes) per 32 edges
    const int edgeBlocks = (E + 255) / 256;   // 8 waves/block * 32 edges/wave = 256 edges/block

    // ---- Layer 1 ----
    k_gemm128<<<(N + 15) / 16, 256, 0, stream>>>(x, W1, bufh, N);
    k_self<<<(n4 + 255) / 256, 256, 0, stream>>>(bufh, dinv, bufa, n4);
    k_edge<<<edgeBlocks, 256, 0, stream>>>(bufh, src, dst, dinv, bufa, E);
    k_bias_relu<<<(n4 + 255) / 256, 256, 0, stream>>>(bufa, b1, n4);

    // ---- Layer 2 ----  (reuse bufh for h2; then overwrite bufa with agg2)
    k_gemm128<<<(N + 15) / 16, 256, 0, stream>>>(bufa, W2, bufh, N);
    k_self<<<(n4 + 255) / 256, 256, 0, stream>>>(bufh, dinv, bufa, n4);
    k_edge<<<edgeBlocks, 256, 0, stream>>>(bufh, src, dst, dinv, bufa, E);
    k_bias_relu<<<(n4 + 255) / 256, 256, 0, stream>>>(bufa, b2, n4);

    // ---- Head ----
    k_head<<<((size_t)N * 32 + 255) / 256, 256, 0, stream>>>(bufa, Wl, bl, (float*)d_out, N);
}
